// ConsisGAD_79577154060657
// MI455X (gfx1250) — compile-verified, matrix operand order checked by reading, not run on hardware
//
#include <hip/hip_runtime.h>
#include <hip/hip_bf16.h>

// ---------------------------------------------------------------------------
// 2-layer GCN on gfx1250 (MI455X).
//   conv(h, W, b) = segment_sum( (h@W)[row] * norm ) + b,  norm = dinv[row]*dinv[col]
//   out = conv(relu(conv(x,W1,b1)), W2, b2), self-loops included.
// Layer-1 GEMM: v_wmma_f32_16x16x32_f16, x-tile staged via async load-to-LDS.
// Aggregation init (bias + self-loop) fused into GEMM epilogues: saves a full
// 100 MB HBM round trip vs. a separate init pass.
// ---------------------------------------------------------------------------

typedef __attribute__((ext_vector_type(16))) _Float16 v16h;
typedef __attribute__((ext_vector_type(8)))  float    v8f;
typedef int v4i_ __attribute__((vector_size(16)));     // i32x4 for async b128

#define D 128   // D_IN == D_H == 128

#define AS1 __attribute__((address_space(1)))
#define AS3 __attribute__((address_space(3)))

#if defined(__has_builtin)
#if __has_builtin(__builtin_amdgcn_global_load_async_to_lds_b128)
#define HAVE_ASYNC_LDS 1
#endif
#endif

__device__ __forceinline__ void atomicAddF(float* p, float v) {
  __hip_atomic_fetch_add(p, v, __ATOMIC_RELAXED, __HIP_MEMORY_SCOPE_AGENT);
}

// ---- degree / normalization ------------------------------------------------
__global__ void __launch_bounds__(256) k_deg_init(float* deg, int N) {
  int i = blockIdx.x * 256 + threadIdx.x;
  if (i < N) deg[i] = 1.0f;                       // self-loop
}

__global__ void __launch_bounds__(256) k_deg_scatter(const int* __restrict__ col,
                                                     float* deg, int E) {
  int e = blockIdx.x * 256 + threadIdx.x;
  if (e < E) atomicAddF(&deg[col[e]], 1.0f);
}

__global__ void __launch_bounds__(256) k_dinv(float* deg, int N) {
  int i = blockIdx.x * 256 + threadIdx.x;
  if (i < N) deg[i] = rsqrtf(deg[i]);             // deg >= 1 always
}

// ---- layer-1 GEMM + fused aggregation init ---------------------------------
// Block = 256 threads = 8 waves; block owns 16 output rows (a contiguous 8 KB
// slab of x), wave w owns columns [16w, 16w+16). K = 128 -> 4 WMMAs per wave.
// Epilogue writes h1 (for the edge gather) and agg = b1 + dinv^2 * h1.
__global__ void __launch_bounds__(256) k_gemm1_wmma(const float* __restrict__ x,
                                                    const float* __restrict__ W1,
                                                    const float* __restrict__ dinv,
                                                    const float* __restrict__ b1,
                                                    float* __restrict__ h1,
                                                    float* __restrict__ agg, int N) {
  __shared__ float    xsf[16 * D];       // x tile, f32, row-major [m][k]  (8 KB)
  __shared__ _Float16 wsT[D * D];        // W1 transposed f16 [n][k]      (32 KB)

  const int tid    = threadIdx.x;
  const int m_base = blockIdx.x * 16;
  const bool full  = (m_base + 16 <= N);

#if HAVE_ASYNC_LDS
  if (full) {
    // Tile rows are consecutive in memory: one contiguous 8192 B region.
    // 256 threads x 2 x b128 async copies, overlapped with W1 staging below.
    const float* gsrc = x + (size_t)m_base * D;
    __builtin_amdgcn_global_load_async_to_lds_b128(
        (AS1 v4i_*)(gsrc + tid * 4),         (AS3 v4i_*)(xsf + tid * 4),         0, 0);
    __builtin_amdgcn_global_load_async_to_lds_b128(
        (AS1 v4i_*)(gsrc + (256 + tid) * 4), (AS3 v4i_*)(xsf + (256 + tid) * 4), 0, 0);
  } else
#endif
  {
    for (int i = tid; i < 16 * D; i += 256) {
      int m = m_base + (i >> 7);
      xsf[i] = (m < N) ? x[(size_t)m * D + (i & (D - 1))] : 0.0f;
    }
  }

  // Stage W1 (transposed, f32->f16). W1 is [k][n] row-major in memory.
  for (int i = tid; i < D * D; i += 256) {
    int k = i >> 7, n = i & (D - 1);
    wsT[n * D + k] = (_Float16)W1[i];
  }

#if HAVE_ASYNC_LDS
  if (full) asm volatile("s_wait_asynccnt 0" ::: "memory");
#endif
  __syncthreads();

  const int wave   = tid >> 5;
  const int lane   = tid & 31;
  const int half   = lane >> 4;    // lane group 0-15 vs 16-31
  const int mn     = lane & 15;    // A row / B,C column within tile
  const int n_base = wave * 16;

  v8f c = {};
  for (int s = 0; s < 4; ++s) {    // K slabs of 32
    v16h a, b;
#pragma unroll
    for (int v = 0; v < 8; ++v) {
      // 16-bit A 16x32 VGPR layout (ISA 7.12.2): VGPR v holds K pair
      //   (v%4)*2 + (v/4)*16 + half*8 ; B mirrors with column index.
      int kk = s * 32 + (v & 3) * 2 + (v >> 2) * 16 + half * 8;
      a[2 * v]     = (_Float16)xsf[mn * D + kk];
      a[2 * v + 1] = (_Float16)xsf[mn * D + kk + 1];
      b[2 * v]     = wsT[(n_base + mn) * D + kk];
      b[2 * v + 1] = wsT[(n_base + mn) * D + kk + 1];
    }
    c = __builtin_amdgcn_wmma_f32_16x16x32_f16(false, a, false, b,
                                               (short)0, c, false, false);
  }

  // 32-bit C/D 16x16 layout: VGPR v -> row (v + 8*half), col mn.
  // Lanes 0-15 / 16-31 each cover 64 contiguous bytes of a row per VGPR.
  const float bn = b1[n_base + mn];
  if (full) {
#pragma unroll
    for (int v = 0; v < 8; ++v) {
      int m = m_base + v + half * 8;
      float d = dinv[m];
      size_t o = (size_t)m * D + n_base + mn;
      h1[o]  = c[v];
      agg[o] = bn + d * d * c[v];
    }
  } else {
#pragma unroll
    for (int v = 0; v < 8; ++v) {
      int m = m_base + v + half * 8;
      if (m < N) {
        float d = dinv[m];
        size_t o = (size_t)m * D + n_base + mn;
        h1[o]  = c[v];
        agg[o] = bn + d * d * c[v];
      }
    }
  }
}

// ---- layer-1 edge scatter: wave per edge, float4 per lane ------------------
// One wave issues a contiguous 512 B gather + 512 B span of f32 atomics.
__global__ void __launch_bounds__(256) k_scatter1(const int* __restrict__ row,
                                                  const int* __restrict__ col,
                                                  const float* __restrict__ h1,
                                                  const float* __restrict__ dinv,
                                                  float* __restrict__ agg, int E) {
  long long gt = (long long)blockIdx.x * 256 + threadIdx.x;
  int e    = (int)(gt >> 5);
  int lane = (int)(gt & 31);
  if (e >= E) return;
  int r = row[e], c = col[e];
  __builtin_prefetch(&h1[(size_t)r * D], 0, 0);        // global_prefetch_b8
  float w = dinv[r] * dinv[c];
  float4 m = ((const float4*)(h1 + (size_t)r * D))[lane];
  float* dst = agg + (size_t)c * D + lane * 4;
  atomicAddF(dst + 0, w * m.x);
  atomicAddF(dst + 1, w * m.y);
  atomicAddF(dst + 2, w * m.z);
  atomicAddF(dst + 3, w * m.w);
}

// ---- fused ReLU + layer-2 GEMM (N-dim = 2) + out init: wave per node -------
__global__ void __launch_bounds__(256) k_gemm2(const float* __restrict__ agg,
                                               const float* __restrict__ W2,
                                               const float* __restrict__ dinv,
                                               const float* __restrict__ b2,
                                               float* __restrict__ h2,
                                               float* __restrict__ out, int N) {
  int tid  = threadIdx.x;
  int node = blockIdx.x * 8 + (tid >> 5);
  int lane = tid & 31;
  if (node >= N) return;
  float4 v = ((const float4*)(agg + (size_t)node * D))[lane];
  v.x = fmaxf(v.x, 0.0f); v.y = fmaxf(v.y, 0.0f);
  v.z = fmaxf(v.z, 0.0f); v.w = fmaxf(v.w, 0.0f);
  const float2* W2v = (const float2*)W2;               // W2 is [k][2]
  int k = lane * 4;
  float2 w0 = W2v[k + 0], w1 = W2v[k + 1], w2 = W2v[k + 2], w3 = W2v[k + 3];
  float a0 = v.x * w0.x + v.y * w1.x + v.z * w2.x + v.w * w3.x;
  float a1 = v.x * w0.y + v.y * w1.y + v.z * w2.y + v.w * w3.y;
#pragma unroll
  for (int off = 16; off > 0; off >>= 1) {
    a0 += __shfl_down(a0, off, 32);
    a1 += __shfl_down(a1, off, 32);
  }
  if (lane == 0) {
    float d = dinv[node], dd = d * d;
    h2[(size_t)node * 2]      = a0;
    h2[(size_t)node * 2 + 1]  = a1;
    out[(size_t)node * 2]     = b2[0] + dd * a0;   // bias + self-loop term
    out[(size_t)node * 2 + 1] = b2[1] + dd * a1;
  }
}

__global__ void __launch_bounds__(256) k_scatter2(const int* __restrict__ row,
                                                  const int* __restrict__ col,
                                                  const float* __restrict__ h2,
                                                  const float* __restrict__ dinv,
                                                  float* __restrict__ out, int E) {
  int e = blockIdx.x * 256 + threadIdx.x;
  if (e >= E) return;
  int r = row[e], c = col[e];
  float w = dinv[r] * dinv[c];
  atomicAddF(&out[(size_t)c * 2 + 0], w * h2[(size_t)r * 2 + 0]);
  atomicAddF(&out[(size_t)c * 2 + 1], w * h2[(size_t)r * 2 + 1]);
}

// ---------------------------------------------------------------------------
extern "C" void kernel_launch(void* const* d_in, const int* in_sizes, int n_in,
                              void* d_out, int out_size, void* d_ws, size_t ws_size,
                              hipStream_t stream) {
  const float* x  = (const float*)d_in[0];
  const int*   ei = (const int*)d_in[1];       // [2, E]: row then col
  const float* W1 = (const float*)d_in[2];
  const float* b1 = (const float*)d_in[3];
  const float* W2 = (const float*)d_in[4];
  const float* b2 = (const float*)d_in[5];
  float* out = (float*)d_out;

  const int N = in_sizes[0] / D;
  const int E = in_sizes[1] / 2;
  const int* row = ei;
  const int* col = ei + E;

  // workspace layout (floats): dinv | h1 | agg1 | h2
  const size_t NP = ((size_t)N + 255) & ~(size_t)255;
  float* wsf  = (float*)d_ws;
  float* dinv = wsf;
  float* h1   = wsf + NP;
  float* agg1 = h1 + NP * D;
  float* h2   = agg1 + NP * D;
  (void)ws_size; (void)n_in; (void)out_size;

  const int nb_n  = (N + 255) / 256;
  const int nb_e  = (E + 255) / 256;
  const int nb_g1 = (N + 15) / 16;
  const int nb_s1 = (int)(((long long)E * 32 + 255) / 256);
  const int nb_g2 = (N + 7) / 8;

  k_deg_init   <<<nb_n,  256, 0, stream>>>(dinv, N);
  k_deg_scatter<<<nb_e,  256, 0, stream>>>(col, dinv, E);
  k_dinv       <<<nb_n,  256, 0, stream>>>(dinv, N);
  k_gemm1_wmma <<<nb_g1, 256, 0, stream>>>(x, W1, dinv, b1, h1, agg1, N);
  k_scatter1   <<<nb_s1, 256, 0, stream>>>(row, col, h1, dinv, agg1, E);
  k_gemm2      <<<nb_g2, 256, 0, stream>>>(agg1, W2, dinv, b2, h2, out, N);
  k_scatter2   <<<nb_e,  256, 0, stream>>>(row, col, h2, dinv, out, E);
}
